// LSTMLayer_89575837926117
// MI455X (gfx1250) — matrix operand order repaired and checked by reading, not compiled, not run on hardware
//
#include <hip/hip_runtime.h>

// Problem constants (fixed by the reference).
constexpr int Bb = 64;     // batch
constexpr int Ls = 2048;   // sequence length
constexpr int Dd = 256;    // input dim (K for forward GEMM)
constexpr int Hh = 256;    // hidden dim

typedef __attribute__((ext_vector_type(16))) __bf16 v16bf;
typedef __attribute__((ext_vector_type(8)))  float  v8f;

// Fragment sizes: per (gate, ntile, kchunk): 32 lanes x 16 bf16 elements.
// One weight family (3 gates): 3 * 16 ntiles * 8 kchunks * 32 * 16 = 196608 bf16.
constexpr size_t FRAG_FAMILY_ELEMS = 3ull * 16 * 8 * 32 * 16; // 196608

// ---------------------------------------------------------------------------
// Prep: convert f32 weights [K][N] into bf16 WMMA B-fragment layout.
// B-fragment (16x16x32 bf16): lane l (<16): N = nt*16 + l, K = kc*32 + e (e<8), kc*32 + 8 + e (e>=8)
//                             lane l (>=16): same N, K = kc*32 + 8 + e (e<8), kc*32 + 16 + e (e>=8)
// dst index = ((gate*16 + nt)*8 + kc)*512 + lane*16 + e  == linear r below.
// ---------------------------------------------------------------------------
__global__ __launch_bounds__(256) void prep_kernel(
    const float* __restrict__ Wf0, const float* __restrict__ Wf1, const float* __restrict__ Wf2,
    const float* __restrict__ Wr0, const float* __restrict__ Wr1, const float* __restrict__ Wr2,
    __bf16* __restrict__ frag)  // [2 families][196608]
{
    const int r   = blockIdx.x * 256 + threadIdx.x;  // 0..196607
    const int fam = blockIdx.y;                      // 0=fwd, 1=rec
    const int e    = r & 15;
    const int lane = (r >> 4) & 31;
    const int kc   = (r >> 9) & 7;
    const int nt   = (r >> 12) & 15;
    const int gate = r >> 16;                        // 0..2
    const int n = nt * 16 + (lane & 15);
    const int k = kc * 32 + ((lane & 16) ? 8 : 0) + (e < 8 ? e : e + 8);
    const float* W = (fam == 0)
        ? (gate == 0 ? Wf0 : (gate == 1 ? Wf1 : Wf2))
        : (gate == 0 ? Wr0 : (gate == 1 ? Wr1 : Wr2));
    frag[(size_t)fam * FRAG_FAMILY_ELEMS + r] = (__bf16)W[(size_t)k * Hh + n];
}

// ---------------------------------------------------------------------------
// Phase 1: pre-activations  xp[t][gate][mtile][ntile] = bias + x_t @ W_fwd
// One wave per (t, gate, mtile) strip: 16 ntiles, K=256 in 8 chunks of 32.
// All 16 B-fragments of a k-chunk are preloaded (32 b128 loads in flight),
// A row data double-buffered across k-chunks.
// Output stored bf16 in C-fragment layout: [tile][lane][8], 16 B per lane.
// ---------------------------------------------------------------------------
__global__ __launch_bounds__(256) void proj_kernel(
    const float* __restrict__ x,
    const float* __restrict__ bci, const float* __restrict__ big, const float* __restrict__ bog,
    const __bf16* __restrict__ Wfwd_frag, __bf16* __restrict__ xp)
{
    const int wave = threadIdx.x >> 5;
    const int lane = threadIdx.x & 31;
    const int strip = blockIdx.x * 8 + wave;         // 0 .. 24575 (L*3*4)
    const int t = strip / 12;
    const int rem = strip % 12;
    const int gate = rem >> 2;
    const int mtile = rem & 3;

    const int row = mtile * 16 + (lane & 15);        // batch row
    const float* xrow = x + ((size_t)row * Ls + t) * Dd;
    const int khalf = (lane & 16) ? 8 : 0;           // A-frag K split per half-wave
    const float* bp = (gate == 0) ? bci : ((gate == 1) ? big : bog);

    v8f acc[16];
#pragma unroll
    for (int nt = 0; nt < 16; ++nt) {
        const float bv = bp[nt * 16 + (lane & 15)];
#pragma unroll
        for (int i = 0; i < 8; ++i) acc[nt][i] = bv;
    }

    const v16bf* Bbase = (const v16bf*)Wfwd_frag + (size_t)gate * 16 * 8 * 32;

    // A raw data: f[0..7] = xrow[kbase..+7], f[8..15] = xrow[kbase+16..+23]
    union Raw { float4 f4[4]; float f[16]; } raw[2];
    {
        const int kbase = khalf;
        raw[0].f4[0] = *(const float4*)(xrow + kbase);
        raw[0].f4[1] = *(const float4*)(xrow + kbase + 4);
        raw[0].f4[2] = *(const float4*)(xrow + kbase + 16);
        raw[0].f4[3] = *(const float4*)(xrow + kbase + 20);
    }

#pragma unroll
    for (int kc = 0; kc < 8; ++kc) {
        const int cur = kc & 1;
        if (kc < 7) {                                 // issue next chunk's A loads early
            const int kbase = (kc + 1) * 32 + khalf;
            raw[cur ^ 1].f4[0] = *(const float4*)(xrow + kbase);
            raw[cur ^ 1].f4[1] = *(const float4*)(xrow + kbase + 4);
            raw[cur ^ 1].f4[2] = *(const float4*)(xrow + kbase + 16);
            raw[cur ^ 1].f4[3] = *(const float4*)(xrow + kbase + 20);
        }
        union { v16bf v; __bf16 e[16]; } a;
#pragma unroll
        for (int i = 0; i < 16; ++i) a.e[i] = (__bf16)raw[cur].f[i];

        v16bf bf[16];
#pragma unroll
        for (int nt = 0; nt < 16; ++nt)
            bf[nt] = Bbase[((size_t)nt * 8 + kc) * 32 + lane];
#pragma unroll
        for (int nt = 0; nt < 16; ++nt)
            acc[nt] = __builtin_amdgcn_wmma_f32_16x16x32_bf16(
                false, a.v, false, bf[nt], (short)0, acc[nt], false, false);
    }

#pragma unroll
    for (int nt = 0; nt < 16; ++nt) {
        union { __bf16 e[8]; uint4 u; } p;
#pragma unroll
        for (int i = 0; i < 8; ++i) p.e[i] = (__bf16)acc[nt][i];
        const size_t tile = (((size_t)t * 3 + gate) * 4 + mtile) * 16 + nt;
        *(uint4*)(xp + tile * 256 + lane * 8) = p.u;
    }
}

// ---------------------------------------------------------------------------
// Phase 2: sequential recurrence. 4 blocks x 512 threads (16 waves), each
// block owns 16 batch rows. Wave w owns N-tile w for ALL THREE gates, so the
// gate combine (c/h update) happens entirely in registers; cell state c is
// register-resident f32 across all 2048 steps. W_rec fragments: kc=0..6 in
// registers (168 VGPRs), kc=7 in LDS. No global loads in the k-loop.
// ---------------------------------------------------------------------------
__global__ __launch_bounds__(512) void rec_kernel(
    const __bf16* __restrict__ Wrec_frag, const __bf16* __restrict__ xp,
    const float* __restrict__ h0, const float* __restrict__ c0,
    float* __restrict__ out)
{
    __shared__ __bf16 h_lds[16][264];             // +8 bf16 pad for A-frag reads
    __shared__ __bf16 bT_lds[3][16][32][16];      // kc=7 B-fragments, 48 KB

    const int wave = threadIdx.x >> 5;            // == ntile
    const int lane = threadIdx.x & 31;
    const int nt   = wave;
    const int mbase = blockIdx.x * 16;
    const int nlo  = lane & 15;
    const int n    = nt * 16 + nlo;
    const int moff  = (lane & 16) ? 8 : 0;
    const int khalf = (lane & 16) ? 8 : 0;
    const int mrow  = lane & 15;

    // h init
    for (int idx = threadIdx.x; idx < 16 * 256; idx += 512) {
        const int m = idx >> 8, nn = idx & 255;
        h_lds[m][nn] = (__bf16)h0[nn];
    }
    // c state: register-resident, c0 broadcast over batch rows
    v8f cst;
    {
        const float cv = c0[n];
#pragma unroll
        for (int r = 0; r < 8; ++r) cst[r] = cv;
    }

    // Preload W_rec B-fragments: kc 0..6 -> registers, kc 7 -> LDS.
    const v16bf* Bb16 = (const v16bf*)Wrec_frag;
    v16bf barr[3][7];
#pragma unroll
    for (int g = 0; g < 3; ++g) {
#pragma unroll
        for (int kc = 0; kc < 7; ++kc)
            barr[g][kc] = Bb16[(((size_t)g * 16 + nt) * 8 + kc) * 32 + lane];
        *(v16bf*)&bT_lds[g][nt][lane][0] =
            Bb16[(((size_t)g * 16 + nt) * 8 + 7) * 32 + lane];
    }
    __syncthreads();

    for (int t = 0; t < Ls; ++t) {
        if (t + 1 < Ls) {
#pragma unroll
            for (int g = 0; g < 3; ++g) {
                const size_t tile = (((size_t)(t + 1) * 3 + g) * 4 + blockIdx.x) * 16 + nt;
                __builtin_prefetch(xp + tile * 256 + lane * 8, 0, 0);
            }
        }

        v8f acc[3];
#pragma unroll
        for (int g = 0; g < 3; ++g)
#pragma unroll
            for (int i = 0; i < 8; ++i) acc[g][i] = 0.0f;

#pragma unroll
        for (int kc = 0; kc < 8; ++kc) {
            union { v16bf v; __bf16 e[16]; } a;
            const int kbase = kc * 32 + khalf;
#pragma unroll
            for (int i = 0; i < 8; ++i) {
                a.e[i]     = h_lds[mrow][kbase + i];
                a.e[i + 8] = h_lds[mrow][kbase + 16 + i];
            }
#pragma unroll
            for (int g = 0; g < 3; ++g) {
                const v16bf bf = (kc < 7) ? barr[g][kc]
                                          : *(const v16bf*)&bT_lds[g][nt][lane][0];
                acc[g] = __builtin_amdgcn_wmma_f32_16x16x32_bf16(
                    false, a.v, false, bf, (short)0, acc[g], false, false);
            }
        }

        // Fold in pre-activations (prefetched last step -> cache-hot).
#pragma unroll
        for (int g = 0; g < 3; ++g) {
            const size_t tile = (((size_t)t * 3 + g) * 4 + blockIdx.x) * 16 + nt;
            union { uint4 u; __bf16 e[8]; } xg;
            xg.u = *(const uint4*)(xp + tile * 256 + lane * 8);
#pragma unroll
            for (int i = 0; i < 8; ++i) acc[g][i] += (float)xg.e[i];
        }

        // In-register gate combine: this wave owns ci/ig/og for the same (m,n).
        __bf16 hreg[8];
#pragma unroll
        for (int r = 0; r < 8; ++r) {
            const float ci = tanhf(acc[0][r]);
            const float ig = 1.0f / (1.0f + __expf(-acc[1][r]));
            const float og = 1.0f / (1.0f + __expf(-acc[2][r]));
            const float c  = cst[r] + ci * ig;     // forget gate cut -> fg == 1
            cst[r] = c;
            const float h = tanhf(c) * og;
            hreg[r] = (__bf16)h;
            if (t == Ls - 1) out[(size_t)(mbase + r + moff) * Hh + n] = h;
        }

        __syncthreads();                            // all k-loop reads of h_lds done
#pragma unroll
        for (int r = 0; r < 8; ++r) h_lds[r + moff][n] = hreg[r];
        __syncthreads();                            // new h visible to all waves
    }
}

// ---------------------------------------------------------------------------
extern "C" void kernel_launch(void* const* d_in, const int* in_sizes, int n_in,
                              void* d_out, int out_size, void* d_ws, size_t ws_size,
                              hipStream_t stream) {
    const float* x   = (const float*)d_in[0];
    const float* Wf0 = (const float*)d_in[1];
    const float* Wf1 = (const float*)d_in[2];
    const float* Wf2 = (const float*)d_in[3];
    const float* Wr0 = (const float*)d_in[4];
    const float* Wr1 = (const float*)d_in[5];
    const float* Wr2 = (const float*)d_in[6];
    const float* bci = (const float*)d_in[7];
    const float* big = (const float*)d_in[8];
    const float* bog = (const float*)d_in[9];
    const float* h0  = (const float*)d_in[10];
    const float* c0  = (const float*)d_in[11];

    // Workspace layout (bf16 elements):
    //   [0)            Wfwd fragments : 196608
    //   [196608)       Wrec fragments : 196608
    //   [393216)       xp pre-activations: L*3*4*16*256 = 100,663,296  (~192 MB)
    __bf16* Wfwd = (__bf16*)d_ws;
    __bf16* Wrec = Wfwd + FRAG_FAMILY_ELEMS;
    __bf16* xpm  = Wrec + FRAG_FAMILY_ELEMS;

    prep_kernel<<<dim3(768, 2), 256, 0, stream>>>(Wf0, Wf1, Wf2, Wr0, Wr1, Wr2, Wfwd);
    proj_kernel<<<3072, 256, 0, stream>>>(x, bci, big, bog, Wfwd, xpm);
    rec_kernel<<<4, 512, 0, stream>>>(Wrec, xpm, h0, c0, (float*)d_out);
}
